// TimingMeta_41704132444581
// MI455X (gfx1250) — matrix operand order repaired and checked by reading, not compiled, not run on hardware
//
#include <hip/hip_runtime.h>
#include <hip/hip_bf16.h>

// ---------------------------------------------------------------------------
// 2-layer LSTM (H=1024, B=64, T=256, D_IN=257) for MI455X (gfx1250, wave32).
//
//   * bf16 weights/activations (range-safe: U(+-1/32) weights, (-1,1) acts),
//     f32 accumulation via v_wmma_f32_16x16x32_bf16.
//   * ONE persistent kernel per layer: loops over all T=256 steps internally.
//     Per step: WMMA GEMM phase (gates = [x_t | h] @ [Wih|Whh]^T, W streamed
//     once from L2) -> software grid barrier -> fused LSTM cell phase ->
//     barrier. Avoids ~1500 sequential kernel-launch nodes (which would cost
//     more than the ~451 GFLOP of math itself).
//   * Layer 1 cell fuses tanh + Wout projection (LDS reduction) -> d_out.
// ---------------------------------------------------------------------------

typedef __bf16 bf16_t;
typedef __attribute__((ext_vector_type(16))) __bf16 v16bf;
typedef __attribute__((ext_vector_type(8)))  __bf16 v8bf;
typedef __attribute__((ext_vector_type(8)))  float   v8f;

#define H_    1024
#define FOURH 4096
#define B_    64
#define T_    256
#define DRAW  257
#define DINP  288        // 257 padded to a multiple of 32 (bf16 WMMA K-chunk)
#define NBLK  64         // persistent grid: 64 blocks x 128 threads (4 waves)

__device__ __forceinline__ float sigmoidf_(float x) {
  return 1.0f / (1.0f + __expf(-x));
}

// A fragment (16x32 bf16, M x K), contiguous K order (B uses the same):
//  lanes 0-15 : row m0+lane,    K = k0+0..15
//  lanes 16-31: row m0+lane-16, K = k0+16..31
__device__ __forceinline__ v16bf load_a(const bf16_t* __restrict__ A, int lda,
                                        int m0, int k0, int mrow, int sel) {
  const bf16_t* p = A + (size_t)(m0 + mrow) * lda + k0 + sel * 16;
  v8bf lo = *(const v8bf*)p;
  v8bf hi = *(const v8bf*)(p + 8);
  v16bf r;
#pragma unroll
  for (int i = 0; i < 8; ++i) { r[i] = lo[i]; r[i + 8] = hi[i]; }
  return r;
}

// ---- software grid barrier (all NBLK blocks co-resident by construction) ----
__device__ __forceinline__ void grid_sync(unsigned* __restrict__ cnt,
                                          unsigned* __restrict__ gen) {
  __threadfence();                       // release this thread's stores (agent)
  __syncthreads();
  if (threadIdx.x == 0) {
    unsigned g = __hip_atomic_load(gen, __ATOMIC_RELAXED, __HIP_MEMORY_SCOPE_AGENT);
    unsigned arrived =
        __hip_atomic_fetch_add(cnt, 1u, __ATOMIC_ACQ_REL, __HIP_MEMORY_SCOPE_AGENT) + 1u;
    if (arrived == (unsigned)NBLK) {
      __hip_atomic_store(cnt, 0u, __ATOMIC_RELAXED, __HIP_MEMORY_SCOPE_AGENT);
      __hip_atomic_fetch_add(gen, 1u, __ATOMIC_RELEASE, __HIP_MEMORY_SCOPE_AGENT);
    } else {
      while (__hip_atomic_load(gen, __ATOMIC_ACQUIRE, __HIP_MEMORY_SCOPE_AGENT) == g) {
        __builtin_amdgcn_s_sleep(2);     // polite spin
      }
    }
  }
  __syncthreads();
  __threadfence();                       // acquire: invalidate stale lines
}

// ---------------------------------------------------------------------------
// Persistent LSTM layer. Grid: 64 blocks x 128 threads.
// GEMM phase : wave owns one 16-wide N strip (64*4*16 = 4096 gate columns);
//              fused K loop over input part (A1 = x_t) and recurrent part
//              (A2 = h). W (row-major N x K) streamed exactly once per step.
// Cell phase : block owns one batch row b = blockIdx.x (1024 columns,
//              8 per thread). IS_L1 additionally does tanh + Wout projection
//              with an LDS tree reduction into out[b,t].
// ---------------------------------------------------------------------------
template <bool IS_L1>
__global__ void __launch_bounds__(128) lstm_layer_persistent(
    const bf16_t* __restrict__ A1base, int lda1, int a1_tstride,
    const bf16_t* __restrict__ Wcat, int ldw, int K1, int K2,
    const float* __restrict__ bias,
    float* __restrict__ cbuf, bf16_t* __restrict__ hbuf,
    bf16_t* __restrict__ yout,                         // layer 0 only
    const float* __restrict__ Wout,                    // layer 1 only
    const float* __restrict__ bout,                    // layer 1 only
    float* __restrict__ out,                           // layer 1 only
    float* __restrict__ gates,
    unsigned* __restrict__ sync) {
  __shared__ float red[128];
  const int tid  = threadIdx.x;
  const int lane = tid & 31;
  const int wave = tid >> 5;                           // 0..3
  const int n0   = blockIdx.x * 64 + wave * 16;
  const int mrow = lane & 15;
  const int sel  = lane >> 4;

  // B fragment source: lane holds W row (= gate column) n0+mrow, K half = sel
  const bf16_t* wp1 = Wcat + (size_t)(n0 + mrow) * ldw + sel * 16;
  const bf16_t* wp2 = wp1 + K1;

  unsigned* cnt = sync;
  unsigned* gen = sync + 64;                           // separate cacheline

  for (int t = 0; t < T_; ++t) {
    // ================= GEMM phase =================
    v8f acc0 = {}, acc1 = {}, acc2 = {}, acc3 = {};
    const bf16_t* A1 = A1base + (size_t)t * a1_tstride;

    for (int k0 = 0; k0 < K1; k0 += 32) {
      v16bf bfrag = *(const v16bf*)(wp1 + k0);
      __builtin_prefetch((const void*)(wp1 + k0 + 128), 0, 1);
      v16bf a0 = load_a(A1, lda1,  0, k0, mrow, sel);
      v16bf a1 = load_a(A1, lda1, 16, k0, mrow, sel);
      v16bf a2 = load_a(A1, lda1, 32, k0, mrow, sel);
      v16bf a3 = load_a(A1, lda1, 48, k0, mrow, sel);
      acc0 = __builtin_amdgcn_wmma_f32_16x16x32_bf16(
          false, a0, false, bfrag, (short)0, acc0, false, false);
      acc1 = __builtin_amdgcn_wmma_f32_16x16x32_bf16(
          false, a1, false, bfrag, (short)0, acc1, false, true);
      acc2 = __builtin_amdgcn_wmma_f32_16x16x32_bf16(
          false, a2, false, bfrag, (short)0, acc2, false, true);
      acc3 = __builtin_amdgcn_wmma_f32_16x16x32_bf16(
          false, a3, false, bfrag, (short)0, acc3, false, true);
    }
    for (int k0 = 0; k0 < K2; k0 += 32) {              // recurrent part: A = h
      v16bf bfrag = *(const v16bf*)(wp2 + k0);
      __builtin_prefetch((const void*)(wp2 + k0 + 128), 0, 1);
      v16bf a0 = load_a(hbuf, H_,  0, k0, mrow, sel);
      v16bf a1 = load_a(hbuf, H_, 16, k0, mrow, sel);
      v16bf a2 = load_a(hbuf, H_, 32, k0, mrow, sel);
      v16bf a3 = load_a(hbuf, H_, 48, k0, mrow, sel);
      acc0 = __builtin_amdgcn_wmma_f32_16x16x32_bf16(
          false, a0, false, bfrag, (short)0, acc0, false, false);
      acc1 = __builtin_amdgcn_wmma_f32_16x16x32_bf16(
          false, a1, false, bfrag, (short)0, acc1, false, true);
      acc2 = __builtin_amdgcn_wmma_f32_16x16x32_bf16(
          false, a2, false, bfrag, (short)0, acc2, false, true);
      acc3 = __builtin_amdgcn_wmma_f32_16x16x32_bf16(
          false, a3, false, bfrag, (short)0, acc3, false, true);
    }

    // C/D layout: VGPR r of lane -> M = mtile + sel*8 + r, N = n0 + mrow
    {
      const int nn = n0 + mrow;
      const int mb = sel * 8;
#pragma unroll
      for (int r = 0; r < 8; ++r) {
        gates[(size_t)(mb + r)      * FOURH + nn] = acc0[r];
        gates[(size_t)(16 + mb + r) * FOURH + nn] = acc1[r];
        gates[(size_t)(32 + mb + r) * FOURH + nn] = acc2[r];
        gates[(size_t)(48 + mb + r) * FOURH + nn] = acc3[r];
      }
    }

    grid_sync(cnt, gen);                 // gates visible to all blocks

    // ================= cell phase =================
    {
      const int b = blockIdx.x;
      const float* g = gates + (size_t)b * FOURH;
      float partial = 0.0f;
#pragma unroll
      for (int q = 0; q < 8; ++q) {
        int j = tid + q * 128;
        float gi = g[j]          + bias[j];
        float gf = g[H_ + j]     + bias[H_ + j];
        float gg = g[2*H_ + j]   + bias[2*H_ + j];
        float go = g[3*H_ + j]   + bias[3*H_ + j];
        int cidx = b * H_ + j;
        float cn = sigmoidf_(gf) * cbuf[cidx] + sigmoidf_(gi) * tanhf(gg);
        float hn = sigmoidf_(go) * tanhf(cn);
        cbuf[cidx] = cn;
        hbuf[cidx] = (bf16_t)hn;
        if (!IS_L1) {
          yout[((size_t)b * T_ + t) * H_ + j] = (bf16_t)hn;
        } else {
          partial += tanhf(hn) * Wout[j];
        }
      }
      if (IS_L1) {
        red[tid] = partial;
        __syncthreads();
        for (int s = 64; s > 0; s >>= 1) {
          if (tid < s) red[tid] += red[tid + s];
          __syncthreads();
        }
        if (tid == 0) out[(size_t)b * T_ + t] = red[0] + bout[0];
      }
    }

    grid_sync(cnt, gen);                 // h/c visible before next step's GEMM
  }
}

// ------------------------------ prep kernels -------------------------------
__global__ void __launch_bounds__(256) prep_X(const float* __restrict__ X,
                                              bf16_t* __restrict__ Xbf) {
  int idx = blockIdx.x * 256 + threadIdx.x;     // over B*T*DINP
  int row = idx / DINP;
  int j   = idx - row * DINP;
  float v = (j < DRAW) ? X[(size_t)row * DRAW + j] : 0.0f;
  Xbf[idx] = (bf16_t)v;
}

// W0cat[4096][1312]: cols [0,288)=Wih0 (zero-padded 257->288), [288,1312)=Whh0
__global__ void __launch_bounds__(256) prep_W0(const float* __restrict__ Wih,
                                               const float* __restrict__ Whh,
                                               bf16_t* __restrict__ Wc) {
  int idx = blockIdx.x * 256 + threadIdx.x;     // over 4096*1312
  int n = idx / (DINP + H_);
  int k = idx - n * (DINP + H_);
  float v;
  if (k < DRAW)       v = Wih[(size_t)n * DRAW + k];
  else if (k < DINP)  v = 0.0f;
  else                v = Whh[(size_t)n * H_ + (k - DINP)];
  Wc[idx] = (bf16_t)v;
}

// W1cat[4096][2048]: cols [0,1024)=Wih1, [1024,2048)=Whh1
__global__ void __launch_bounds__(256) prep_W1(const float* __restrict__ Wih,
                                               const float* __restrict__ Whh,
                                               bf16_t* __restrict__ Wc) {
  int idx = blockIdx.x * 256 + threadIdx.x;     // over 4096*2048
  int n = idx >> 11;
  int k = idx & 2047;
  float v = (k < H_) ? Wih[(size_t)n * H_ + k]
                     : Whh[(size_t)n * H_ + (k - H_)];
  Wc[idx] = (bf16_t)v;
}

__global__ void __launch_bounds__(256) prep_bias(const float* __restrict__ a,
                                                 const float* __restrict__ b,
                                                 float* __restrict__ o) {
  int i = blockIdx.x * 256 + threadIdx.x;
  o[i] = a[i] + b[i];
}

__global__ void __launch_bounds__(256) prep_state(const float* __restrict__ h0,
                                                  const float* __restrict__ c0,
                                                  bf16_t* __restrict__ hb,
                                                  float* __restrict__ cb,
                                                  int layer) {
  int i = blockIdx.x * 256 + threadIdx.x;       // over B*H
  hb[i] = (bf16_t)h0[layer * (B_ * H_) + i];
  cb[i] = c0[layer * (B_ * H_) + i];
}

__global__ void init_sync(unsigned* __restrict__ s) {
  if (threadIdx.x < 128) s[threadIdx.x] = 0u;
}

// ---------------------------------------------------------------------------
extern "C" void kernel_launch(void* const* d_in, const int* in_sizes, int n_in,
                              void* d_out, int out_size, void* d_ws, size_t ws_size,
                              hipStream_t stream) {
  const float* X    = (const float*)d_in[0];
  /* d_in[1] = X_lengths : unused by the reference computation */
  const float* h0   = (const float*)d_in[2];
  const float* c0   = (const float*)d_in[3];
  const float* Wih0 = (const float*)d_in[4];
  const float* Whh0 = (const float*)d_in[5];
  const float* bih0 = (const float*)d_in[6];
  const float* bhh0 = (const float*)d_in[7];
  const float* Wih1 = (const float*)d_in[8];
  const float* Whh1 = (const float*)d_in[9];
  const float* bih1 = (const float*)d_in[10];
  const float* bhh1 = (const float*)d_in[11];
  const float* Wout = (const float*)d_in[12];
  const float* bout = (const float*)d_in[13];
  float* out = (float*)d_out;
  (void)in_sizes; (void)n_in; (void)out_size; (void)ws_size;

  // ---- workspace partition (all 256B aligned) ----
  char* base = (char*)d_ws;
  size_t off = 0;
  auto wsalloc = [&](size_t bytes) -> void* {
    void* p = base + off;
    off = (off + bytes + 255) & ~(size_t)255;
    return p;
  };
  bf16_t*   Xbf   = (bf16_t*)wsalloc((size_t)B_ * T_ * DINP * 2);       // 9.4 MB
  bf16_t*   W0cat = (bf16_t*)wsalloc((size_t)FOURH * (DINP + H_) * 2);  // 10.7 MB
  bf16_t*   W1cat = (bf16_t*)wsalloc((size_t)FOURH * (2 * H_) * 2);     // 16.8 MB
  bf16_t*   y1    = (bf16_t*)wsalloc((size_t)B_ * T_ * H_ * 2);         // 33.5 MB
  float*    bias0 = (float*)wsalloc(FOURH * 4);
  float*    bias1 = (float*)wsalloc(FOURH * 4);
  bf16_t*   h0b   = (bf16_t*)wsalloc(B_ * H_ * 2);
  bf16_t*   h1b   = (bf16_t*)wsalloc(B_ * H_ * 2);
  float*    c0b   = (float*)wsalloc(B_ * H_ * 4);
  float*    c1b   = (float*)wsalloc(B_ * H_ * 4);
  float*    gates = (float*)wsalloc((size_t)B_ * FOURH * 4);            // 1 MB
  unsigned* sync  = (unsigned*)wsalloc(128 * sizeof(unsigned));

  // ---- one-time bf16 conversion / packing (fully parallel) ----
  prep_X <<<(B_ * T_ * DINP) / 256, 256, 0, stream>>>(X, Xbf);
  prep_W0<<<(FOURH * (DINP + H_)) / 256, 256, 0, stream>>>(Wih0, Whh0, W0cat);
  prep_W1<<<(FOURH * 2 * H_) / 256, 256, 0, stream>>>(Wih1, Whh1, W1cat);
  prep_bias<<<FOURH / 256, 256, 0, stream>>>(bih0, bhh0, bias0);
  prep_bias<<<FOURH / 256, 256, 0, stream>>>(bih1, bhh1, bias1);
  prep_state<<<(B_ * H_) / 256, 256, 0, stream>>>(h0, c0, h0b, c0b, 0);
  prep_state<<<(B_ * H_) / 256, 256, 0, stream>>>(h0, c0, h1b, c1b, 1);
  init_sync<<<1, 128, 0, stream>>>(sync);

  // ---- layer 0: one persistent kernel scans all T steps ----
  lstm_layer_persistent<false><<<NBLK, 128, 0, stream>>>(
      Xbf, T_ * DINP, DINP,
      W0cat, DINP + H_, DINP, H_,
      bias0, c0b, h0b,
      /*yout=*/y1, /*Wout=*/nullptr, /*bout=*/nullptr, /*out=*/nullptr,
      gates, sync);

  // ---- layer 1: persistent scan + fused tanh/Wout projection -> d_out ----
  lstm_layer_persistent<true><<<NBLK, 128, 0, stream>>>(
      y1, T_ * H_, H_,
      W1cat, 2 * H_, H_, H_,
      bias1, c1b, h1b,
      /*yout=*/nullptr, Wout, bout, out,
      gates, sync);
}
